// GroupedQueryAttention_8117488189629
// MI455X (gfx1250) — matrix operand order repaired and checked by reading, not compiled, not run on hardware
//
#include <hip/hip_runtime.h>
#include <hip/hip_bf16.h>

// ---------------------------------------------------------------------------
// GQA forward for MI455X (gfx1250, wave32, WMMA bf16 w/ f32 accumulate)
// B=1, S=2048, D=128, H=32, KVH=8, REP=4
// ---------------------------------------------------------------------------

typedef __bf16 bf16_t;
typedef __attribute__((ext_vector_type(16))) __bf16 v16bf;
typedef __attribute__((ext_vector_type(8)))  __bf16 v8bf;
typedef __attribute__((ext_vector_type(8)))  float  v8f;

#define SEQ   2048
#define DH    128
#define NH    32
#define NKVH  8

// ---- WMMA helpers ----------------------------------------------------------

__device__ __forceinline__ v8f wmma_bf16(v16bf a, v16bf b, v8f c) {
  // D = A(16x32 bf16) x B(32x16 bf16) + C(16x16 f32)
  return __builtin_amdgcn_wmma_f32_16x16x32_bf16(
      /*neg_a=*/false, a, /*neg_b=*/false, b,
      /*c_mod=*/(short)0, c, /*reuse_a=*/false, /*reuse_b=*/false);
}

// A-matrix fragment (16-bit, 16x32 MxK). Lane l holds row m=l%16.
// half = l/16: element s <- K = 16*half + s (16 contiguous K, matches B).
// `base` points at this lane's row, start of the 32-wide K chunk.
__device__ __forceinline__ v16bf load_a_frag(const bf16_t* base, int half) {
  return *(const v16bf*)(base + 16 * half);
}

// B-matrix fragment (32x16 KxN): lane l holds column n=l%16, 16 consecutive K
// starting at K = 16*(l/16). Caller passes pointer to those 16 contiguous bf16.
// (plain 32B load)

// ---- elementwise converts --------------------------------------------------

__global__ void cvt_bf16_kernel(const float* __restrict__ src,
                                bf16_t* __restrict__ dst, int n) {
  int i = blockIdx.x * blockDim.x + threadIdx.x;
  if (i < n) dst[i] = (bf16_t)src[i];
}

// src[K][N] (f32) -> dst[N][K] (bf16)
__global__ void cvt_bf16_tr_kernel(const float* __restrict__ src,
                                   bf16_t* __restrict__ dst, int K, int N) {
  int i = blockIdx.x * blockDim.x + threadIdx.x;
  if (i < K * N) {
    int k = i / N, n = i % N;
    dst[(size_t)n * K + k] = (bf16_t)src[i];
  }
}

// ---- projection GEMM: C[M,N] = A[M,128] * W[128,N] + bias ------------------
// A: bf16 row-major, Bt: bf16 [N][128] (transposed weight), one wave per 16x16 tile.
template <bool OUT_BF16>
__global__ __launch_bounds__(32)
void gemm_proj_kernel(const bf16_t* __restrict__ A, const bf16_t* __restrict__ Bt,
                      const float* __restrict__ bias, int N, float scale,
                      bf16_t* __restrict__ outb, float* __restrict__ outf) {
  const int n0 = blockIdx.x * 16;
  const int m0 = blockIdx.y * 16;
  const int lane = threadIdx.x;
  const int nn = lane & 15, half = lane >> 4;

  const bf16_t* arow = A + (size_t)(m0 + nn) * 128;   // this lane's A row
  const bf16_t* brow = Bt + (size_t)(n0 + nn) * 128;  // this lane's B column

  v8f acc = {};
#pragma unroll
  for (int k0 = 0; k0 < 128; k0 += 32) {
    v16bf af = load_a_frag(arow + k0, half);
    v16bf bfrag = *(const v16bf*)(brow + k0 + 16 * half);
    acc = wmma_bf16(af, bfrag, acc);
  }

  const float bval = bias[n0 + nn];
#pragma unroll
  for (int j = 0; j < 8; ++j) {
    const int m = m0 + j + 8 * half;
    if (OUT_BF16)
      outb[(size_t)m * N + n0 + nn] = (bf16_t)((acc[j] + bval) * scale);
    else
      outf[(size_t)m * N + n0 + nn] = acc[j] + bval;
  }
}

// ---- RoPE + head rearrange for K / V ---------------------------------------
// P: [2048][1024] f32 projection (bias already added).
// head element (g,t,d) = P[g*256 + t/8][(t%8)*128 + d], rope position = t.
// transpose==0: dst[g][t][d] (Khead, 128 contiguous d)
// transpose==1: dst[g][d][t] (Vt, 2048 contiguous t)
__global__ void rope_rearrange_kernel(const float* __restrict__ P,
                                      bf16_t* __restrict__ dst, int transpose) {
  int idx = blockIdx.x * blockDim.x + threadIdx.x;  // over NKVH*SEQ*64 = 2^20
  if (idx >= NKVH * SEQ * 64) return;
  const int i = idx & 63;              // freq index 0..63
  const int t = (idx >> 6) & (SEQ - 1);
  const int g = idx >> 17;             // SEQ*64 = 2^17

  const int srow = g * 256 + (t >> 3);
  const int scol = (t & 7) * 128;
  const float x0 = P[(size_t)srow * 1024 + scol + i];
  const float x1 = P[(size_t)srow * 1024 + scol + i + 64];

  // inv_freq = 10000^(-i/64) = exp(-i * ln(10000)/64)
  const float freq = __expf(-(float)i * (9.210340371976184f / 64.0f));
  const float ang = (float)t * freq;
  float sn, cs;
  __sincosf(ang, &sn, &cs);
  const float o0 = x0 * cs - x1 * sn;
  const float o1 = x1 * cs + x0 * sn;

  if (transpose) {
    dst[((size_t)g * 128 + i) * SEQ + t] = (bf16_t)o0;
    dst[((size_t)g * 128 + i + 64) * SEQ + t] = (bf16_t)o1;
  } else {
    dst[((size_t)g * SEQ + t) * 128 + i] = (bf16_t)o0;
    dst[((size_t)g * SEQ + t) * 128 + i + 64] = (bf16_t)o1;
  }
}

// ---- flash attention (one wave = one head x 16 query rows) -----------------
// Qp: bf16 [2048][4096], already scaled by 1/sqrt(128) with bias.
// Kh: bf16 [g][t][128] roped.  Vt: bf16 [g][128][t] roped.
// ctx out: bf16 [s][h*128+d]  (== transpose(0,2,1,3) flatten)
// Key chunk = 64: one online-softmax pass (shuffle reductions + O rescale)
// amortized over 32 WMMAs (16 score + 16 AV).
__global__ __launch_bounds__(32)
void attn_kernel(const bf16_t* __restrict__ Qp, const bf16_t* __restrict__ Kh,
                 const bf16_t* __restrict__ Vt, bf16_t* __restrict__ ctx) {
  __shared__ bf16_t p_lds[16 * 64];

  const int h = blockIdx.y;        // 0..31
  const int s0 = blockIdx.x * 16;  // query tile start
  const int g = h & (NKVH - 1);
  const int lane = threadIdx.x;
  const int nn = lane & 15, half = lane >> 4;

  // Q tile: 16 rows live in ONE Qp row (view-reshape quirk), contiguous cols.
  const bf16_t* qbase =
      Qp + (size_t)(h * 64 + (s0 >> 5)) * 4096 + (size_t)(s0 & 31) * 128 + (size_t)nn * 128;
  v16bf qf[4];
#pragma unroll
  for (int kc = 0; kc < 4; ++kc) qf[kc] = load_a_frag(qbase + kc * 32, half);

  v8f o[8];
#pragma unroll
  for (int i = 0; i < 8; ++i) o[i] = (v8f){};
  float m_r[8], l_r[8];
#pragma unroll
  for (int j = 0; j < 8; ++j) { m_r[j] = -3.0e38f; l_r[j] = 0.0f; }

  const bf16_t* kg = Kh + (size_t)g * SEQ * 128;
  const bf16_t* vg = Vt + (size_t)g * 128 * SEQ;

  for (int t0 = 0; t0 < SEQ; t0 += 64) {
    // ---- scores: 16 x 64 over full d=128 (four 16-wide n-chunks) ----
    v8f sc[4];
#pragma unroll
    for (int nc = 0; nc < 4; ++nc) sc[nc] = (v8f){};
#pragma unroll
    for (int nc = 0; nc < 4; ++nc) {
      const bf16_t* kb = kg + (size_t)(t0 + nc * 16 + nn) * 128 + 16 * half;
#pragma unroll
      for (int kc = 0; kc < 4; ++kc) {
        v16bf b = *(const v16bf*)(kb + kc * 32);
        sc[nc] = wmma_bf16(qf[kc], b, sc[nc]);
      }
    }
    if (t0 + 64 < SEQ) {
      __builtin_prefetch(kg + (size_t)(t0 + 64 + nn) * 128, 0, 1);
      __builtin_prefetch(kg + (size_t)(t0 + 80 + nn) * 128, 0, 1);
      __builtin_prefetch(kg + (size_t)(t0 + 96 + nn) * 128, 0, 1);
      __builtin_prefetch(kg + (size_t)(t0 + 112 + nn) * 128, 0, 1);
    }

    // ---- online softmax (row = j + 8*half, reduce across 16-lane half) ----
    float pa[4][8];
#pragma unroll
    for (int j = 0; j < 8; ++j) {
      float mx = fmaxf(fmaxf(sc[0][j], sc[1][j]), fmaxf(sc[2][j], sc[3][j]));
#pragma unroll
      for (int off = 1; off < 16; off <<= 1) mx = fmaxf(mx, __shfl_xor(mx, off, 32));
      const float mnew = fmaxf(m_r[j], mx);
      const float alpha = __expf(m_r[j] - mnew);
      m_r[j] = mnew;
      float rs = 0.0f;
#pragma unroll
      for (int nc = 0; nc < 4; ++nc) {
        const float p = __expf(sc[nc][j] - mnew);
        pa[nc][j] = p;
        rs += p;
      }
#pragma unroll
      for (int off = 1; off < 16; off <<= 1) rs += __shfl_xor(rs, off, 32);
      l_r[j] = l_r[j] * alpha + rs;
      o[0][j] *= alpha; o[1][j] *= alpha; o[2][j] *= alpha; o[3][j] *= alpha;
      o[4][j] *= alpha; o[5][j] *= alpha; o[6][j] *= alpha; o[7][j] *= alpha;
    }

    // ---- P (C-layout f32) -> LDS -> two A-layout bf16 fragments ----
#pragma unroll
    for (int j = 0; j < 8; ++j) {
      const int m = j + 8 * half;
#pragma unroll
      for (int nc = 0; nc < 4; ++nc)
        p_lds[m * 64 + nc * 16 + nn] = (bf16_t)pa[nc][j];
    }
    __syncthreads();
    v16bf pf0 = load_a_frag(p_lds + nn * 64, half);       // t_local 0..31
    v16bf pf1 = load_a_frag(p_lds + nn * 64 + 32, half);  // t_local 32..63
    __syncthreads();

    // ---- O += P x V : 8 d-chunks of 16, two K=32 steps each ----
#pragma unroll
    for (int dc = 0; dc < 8; ++dc) {
      const bf16_t* vb = vg + (size_t)(dc * 16 + nn) * SEQ + t0 + 16 * half;
      v16bf vf0 = *(const v16bf*)vb;
      v16bf vf1 = *(const v16bf*)(vb + 32);
      o[dc] = wmma_bf16(pf0, vf0, o[dc]);
      o[dc] = wmma_bf16(pf1, vf1, o[dc]);
    }
  }

  // ---- epilogue: normalize, store to ctx[s][h*128+d] as bf16 ----
#pragma unroll
  for (int dc = 0; dc < 8; ++dc) {
#pragma unroll
    for (int j = 0; j < 8; ++j) {
      const int m = j + 8 * half;
      const float val = o[dc][j] / l_r[j];
      ctx[(size_t)(s0 + m) * 4096 + h * 128 + dc * 16 + nn] = (bf16_t)val;
    }
  }
}

// ---- final GEMM: out[2048][128] = ctx[2048][4096] x Wo + bo (f32 out) ------
__global__ __launch_bounds__(32)
void gemm_out_kernel(const bf16_t* __restrict__ A, const bf16_t* __restrict__ Bt,
                     const float* __restrict__ bias, float* __restrict__ out) {
  const int n0 = blockIdx.x * 16;
  const int m0 = blockIdx.y * 16;
  const int lane = threadIdx.x;
  const int nn = lane & 15, half = lane >> 4;

  const bf16_t* arow = A + (size_t)(m0 + nn) * 4096;
  const bf16_t* brow = Bt + (size_t)(n0 + nn) * 4096;

  v8f acc = {};
  for (int k0 = 0; k0 < 4096; k0 += 32) {
    v16bf af = load_a_frag(arow + k0, half);
    v16bf bfrag = *(const v16bf*)(brow + k0 + 16 * half);
    acc = wmma_bf16(af, bfrag, acc);
  }
  const float bval = bias[n0 + nn];
#pragma unroll
  for (int j = 0; j < 8; ++j)
    out[(size_t)(m0 + j + 8 * half) * 128 + n0 + nn] = acc[j] + bval;
}

// ---------------------------------------------------------------------------

extern "C" void kernel_launch(void* const* d_in, const int* in_sizes, int n_in,
                              void* d_out, int out_size, void* d_ws, size_t ws_size,
                              hipStream_t stream) {
  const float* query = (const float*)d_in[0];
  const float* keys = (const float*)d_in[1];
  const float* values = (const float*)d_in[2];
  const float* Wq = (const float*)d_in[3];
  const float* bq = (const float*)d_in[4];
  const float* Wk = (const float*)d_in[5];
  const float* bk = (const float*)d_in[6];
  const float* Wv = (const float*)d_in[7];
  const float* bv = (const float*)d_in[8];
  const float* Wo = (const float*)d_in[9];
  const float* bo = (const float*)d_in[10];
  float* out = (float*)d_out;

  // ---- workspace carve-out (bytes, 256B aligned blocks) ----
  char* w = (char*)d_ws;
  size_t off = 0;
  auto carve = [&](size_t bytes) {
    char* p = w + off;
    off += (bytes + 255) & ~(size_t)255;
    return p;
  };
  bf16_t* q_bf  = (bf16_t*)carve((size_t)SEQ * 128 * 2);       // 0.5 MB
  bf16_t* k_bf  = (bf16_t*)carve((size_t)SEQ * 128 * 2);
  bf16_t* v_bf  = (bf16_t*)carve((size_t)SEQ * 128 * 2);
  bf16_t* wqt   = (bf16_t*)carve((size_t)4096 * 128 * 2);      // 1 MB
  bf16_t* wkt   = (bf16_t*)carve((size_t)1024 * 128 * 2);
  bf16_t* wvt   = (bf16_t*)carve((size_t)1024 * 128 * 2);
  bf16_t* wot   = (bf16_t*)carve((size_t)128 * 4096 * 2);
  bf16_t* Qp    = (bf16_t*)carve((size_t)SEQ * 4096 * 2);      // 16 MB
  float*  Kp    = (float*)carve((size_t)SEQ * 1024 * 4);       // 8 MB
  float*  Vp    = (float*)carve((size_t)SEQ * 1024 * 4);       // 8 MB
  bf16_t* Khead = (bf16_t*)carve((size_t)NKVH * SEQ * 128 * 2);// 4 MB
  bf16_t* Vt    = (bf16_t*)carve((size_t)NKVH * 128 * SEQ * 2);// 4 MB
  bf16_t* ctx   = (bf16_t*)carve((size_t)SEQ * 4096 * 2);      // 16 MB
  (void)ws_size; (void)in_sizes; (void)n_in; (void)out_size;

  const int T = 256;
  // 1) bf16 converts (inputs + transposed weights)
  cvt_bf16_kernel<<<(SEQ * 128 + T - 1) / T, T, 0, stream>>>(query, q_bf, SEQ * 128);
  cvt_bf16_kernel<<<(SEQ * 128 + T - 1) / T, T, 0, stream>>>(keys, k_bf, SEQ * 128);
  cvt_bf16_kernel<<<(SEQ * 128 + T - 1) / T, T, 0, stream>>>(values, v_bf, SEQ * 128);
  cvt_bf16_tr_kernel<<<(128 * 4096 + T - 1) / T, T, 0, stream>>>(Wq, wqt, 128, 4096);
  cvt_bf16_tr_kernel<<<(128 * 1024 + T - 1) / T, T, 0, stream>>>(Wk, wkt, 128, 1024);
  cvt_bf16_tr_kernel<<<(128 * 1024 + T - 1) / T, T, 0, stream>>>(Wv, wvt, 128, 1024);
  cvt_bf16_tr_kernel<<<(4096 * 128 + T - 1) / T, T, 0, stream>>>(Wo, wot, 4096, 128);

  // 2) projections (WMMA). Q: scale by 1/sqrt(128), straight to bf16.
  const float qscale = 0.08838834764831845f;  // 1/sqrt(128)
  gemm_proj_kernel<true><<<dim3(4096 / 16, SEQ / 16), 32, 0, stream>>>(
      q_bf, wqt, bq, 4096, qscale, Qp, nullptr);
  gemm_proj_kernel<false><<<dim3(1024 / 16, SEQ / 16), 32, 0, stream>>>(
      k_bf, wkt, bk, 1024, 1.0f, nullptr, Kp);
  gemm_proj_kernel<false><<<dim3(1024 / 16, SEQ / 16), 32, 0, stream>>>(
      v_bf, wvt, bv, 1024, 1.0f, nullptr, Vp);

  // 3) RoPE + head rearrange (K row-major, V transposed)
  const int nrope = NKVH * SEQ * 64;
  rope_rearrange_kernel<<<(nrope + T - 1) / T, T, 0, stream>>>(Kp, Khead, 0);
  rope_rearrange_kernel<<<(nrope + T - 1) / T, T, 0, stream>>>(Vp, Vt, 1);

  // 4) flash attention: grid = (q-tiles, heads), one wave per block
  attn_kernel<<<dim3(SEQ / 16, NH), 32, 0, stream>>>(Qp, Khead, Vt, ctx);

  // 5) output projection -> f32 d_out
  gemm_out_kernel<<<dim3(128 / 16, SEQ / 16), 32, 0, stream>>>(ctx, wot, bo, out);
}
